// Model_13632226198272
// MI455X (gfx1250) — compile-verified, matrix operand order checked by reading, not run on hardware
//
#include <hip/hip_runtime.h>
#include <stdint.h>

#define D_MODEL   512
#define D_STATE   64
#define D_CONV    4
#define HEADDIM   64
#define D_INNER   1024
#define NHEADS    16
#define CONV_DIM  1152
#define D_IN_PROJ 2192
#define BATCH     2
#define SEQLEN    4096
#define NTOK      (BATCH*SEQLEN)
#define Q         64
#define NCHUNK    (SEQLEN/Q)

typedef __attribute__((ext_vector_type(16))) __bf16 v16bf;
typedef __attribute__((ext_vector_type(8)))  float  v8f;
typedef unsigned short u16;

__device__ __forceinline__ unsigned pack_bf16(float a, float b) {
    union { __bf16 h[2]; unsigned u; } r;
    r.h[0] = (__bf16)a; r.h[1] = (__bf16)b;
    return r.u;
}
__device__ __forceinline__ u16 bf16_bits(float f) {
    union { __bf16 h; u16 u; } r; r.h = (__bf16)f; return r.u;
}
__device__ __forceinline__ v8f wmma_bf16(v16bf a, v16bf b, v8f c) {
    return __builtin_amdgcn_wmma_f32_16x16x32_bf16(false, a, false, b, (short)0, c, false, false);
}
__device__ __forceinline__ v16bf frag_ld(const u16* rowbase, int lh) {
    union { uint4 q[2]; v16bf v; } u;
    u.q[0] = *(const uint4*)(rowbase + lh * 8);
    u.q[1] = *(const uint4*)(rowbase + 16 + lh * 8);
    return u.v;
}
// async copy 16B global -> LDS (ASYNCcnt-tracked, no VGPR round trip)
__device__ __forceinline__ void async_copy_b128(const u16* gsrc, u16* ldst) {
    unsigned l = (unsigned)(uintptr_t)ldst;
    unsigned long long g = (unsigned long long)(uintptr_t)gsrc;
    asm volatile("global_load_async_to_lds_b128 %0, %1, off" :: "v"(l), "v"(g) : "memory");
}
__device__ __forceinline__ void async_wait_all() {
    asm volatile("s_wait_asynccnt 0x0" ::: "memory");
}

// =============== pre-conversion kernels ===============
__global__ __launch_bounds__(256) void f32_to_bf16_kernel(
    const float* __restrict__ src, u16* __restrict__ dst, int n8)   // n8 = n/8
{
    int i = blockIdx.x * blockDim.x + threadIdx.x;
    if (i >= n8) return;
    const float4* s = (const float4*)(src + (size_t)i * 8);
    float4 f0 = s[0], f1 = s[1];
    uint4 p;
    p.x = pack_bf16(f0.x, f0.y); p.y = pack_bf16(f0.z, f0.w);
    p.z = pack_bf16(f1.x, f1.y); p.w = pack_bf16(f1.z, f1.w);
    *(uint4*)(dst + (size_t)i * 8) = p;
}

// W [K][N] f32 -> Wt [N][K] bf16, 32x32 tiles through LDS
__global__ __launch_bounds__(256) void transpose_bf16_kernel(
    const float* __restrict__ W, u16* __restrict__ Wt, int K, int N)
{
    __shared__ float tile[32][33];
    int k0 = blockIdx.y * 32;
    int n0 = blockIdx.x * 32;
    int c  = threadIdx.x & 31;
    int r8 = threadIdx.x >> 5;     // 0..7
#pragma unroll
    for (int rr = 0; rr < 32; rr += 8) {
        int k = k0 + rr + r8, n = n0 + c;
        tile[rr + r8][c] = (k < K && n < N) ? W[(size_t)k * N + n] : 0.0f;
    }
    __syncthreads();
#pragma unroll
    for (int rr = 0; rr < 32; rr += 8) {
        int n = n0 + rr + r8, k = k0 + c;
        if (n < N && k < K) Wt[(size_t)n * K + k] = bf16_bits(tile[c][rr + r8]);
    }
}

// =============== WMMA GEMM: C(MxN,f32) = A(MxK,bf16) * Bt(NxK,bf16)^T ===============
#define BM 64
#define BN 128
#define BK 32
#define LDSW 40

__global__ __launch_bounds__(256) void gemm_bf16wmma(
    const u16* __restrict__ A, const u16* __restrict__ Bt, float* __restrict__ C,
    int M, int N, int K)
{
    __shared__ __align__(16) u16 sA[BM * LDSW];
    __shared__ __align__(16) u16 sB[BN * LDSW];

    const int mbase = blockIdx.y * BM;
    const int nbase = blockIdx.x * BN;
    const int lane = threadIdx.x & 31;
    const int wid  = threadIdx.x >> 5;
    const int wm   = wid & 3;
    const int wn   = wid >> 2;
    const int lm   = lane & 15;
    const int lh   = lane >> 4;

    // staging coordinates (pure 16B copies; bf16 data pre-converted)
    const int arow = threadIdx.x >> 2;            // 0..63
    const int aseg = (threadIdx.x & 3) * 8;       // u16 elements
    const u16* agsrc = A + (size_t)(mbase + arow) * K + aseg;
    u16* aldst = &sA[arow * LDSW + aseg];

    v8f acc[4] = {};

    for (int k0 = 0; k0 < K; k0 += BK) {
        async_copy_b128(agsrc + k0, aldst);
        if (k0 + BK < K) __builtin_prefetch((const void*)(agsrc + k0 + BK), 0, 1);
#pragma unroll
        for (int i = 0; i < 2; i++) {
            int idx = threadIdx.x + i * 256;      // 0..511
            int n   = idx >> 2;
            int seg = (idx & 3) * 8;
            int col = nbase + n; if (col > N - 1) col = N - 1;   // clamp (store guarded)
            async_copy_b128(Bt + (size_t)col * K + k0 + seg, &sB[n * LDSW + seg]);
        }
        async_wait_all();
        __syncthreads();

        v16bf fa = frag_ld(&sA[(wm * 16 + lm) * LDSW], lh);
#pragma unroll
        for (int j = 0; j < 4; j++) {
            v16bf fb = frag_ld(&sB[(wn * 64 + j * 16 + lm) * LDSW], lh);
            acc[j] = wmma_bf16(fa, fb, acc[j]);
        }
        __syncthreads();
    }

#pragma unroll
    for (int j = 0; j < 4; j++) {
        int col = nbase + wn * 64 + j * 16 + lm;
        if (col < N) {
#pragma unroll
            for (int r = 0; r < 8; r++) {
                int row = mbase + wm * 16 + lh * 8 + r;
                C[(size_t)row * N + col] = acc[j][r];
            }
        }
    }
}

// =============== depthwise causal conv1d (k=4) + bias + SiLU ===============
__global__ __launch_bounds__(256) void conv_silu_kernel(
    const float* __restrict__ zxbcdt, const float* __restrict__ conv_w,
    const float* __restrict__ conv_b, float* __restrict__ xBC)
{
    int idx = blockIdx.x * blockDim.x + threadIdx.x;
    if (idx >= NTOK * CONV_DIM) return;
    int c = idx % CONV_DIM;
    int t = idx / CONV_DIM;
    int l = t % SEQLEN;
    float acc = conv_b[c];
#pragma unroll
    for (int j = 0; j < D_CONV; j++) {
        int ll = l - (D_CONV - 1) + j;
        if (ll >= 0)
            acc += conv_w[c * D_CONV + j] *
                   zxbcdt[(size_t)(t - l + ll) * D_IN_PROJ + D_INNER + c];
    }
    xBC[idx] = acc / (1.0f + __expf(-acc));
}

// =============== dt = softplus(raw + bias); dtA = dt * (-exp(A_log)) ===============
__global__ __launch_bounds__(256) void dt_kernel(
    const float* __restrict__ zxbcdt, const float* __restrict__ dt_bias,
    const float* __restrict__ A_log, float* __restrict__ dtb, float* __restrict__ dtAb)
{
    int idx = blockIdx.x * blockDim.x + threadIdx.x;
    if (idx >= NTOK * NHEADS) return;
    int h = idx % NHEADS;
    float x = zxbcdt[(size_t)(idx / NHEADS) * D_IN_PROJ + D_INNER + CONV_DIM + h] + dt_bias[h];
    float dt = (x > 20.0f) ? x : log1pf(__expf(x));
    dtb[idx]  = dt;
    dtAb[idx] = dt * (-__expf(A_log[h]));
}

// =============== chunked SSD scan ===============
#define CLDS 72

__device__ __forceinline__ void chunk_cumsum(
    const float* dtAb, int b, int t0, int h, int tid, float* scum)
{
    if (tid < Q) scum[tid] = dtAb[(size_t)(b * SEQLEN + t0 + tid) * NHEADS + h];
    __syncthreads();
    for (int off = 1; off < Q; off <<= 1) {
        float v = 0.0f;
        if (tid < Q && tid >= off) v = scum[tid - off];
        __syncthreads();
        if (tid < Q) scum[tid] += v;
        __syncthreads();
    }
}

// Phase A: T_c[p,n] = sum_s exp(cumA_end - cumA_s)*dt_s * x_s[p] * B_s[n]
__global__ __launch_bounds__(128) void chunk_state_kernel(
    const float* __restrict__ xBC, const float* __restrict__ dtb,
    const float* __restrict__ dtAb, float* __restrict__ tbuf, float* __restrict__ pcum)
{
    const int blk = blockIdx.x;
    const int h = blk % NHEADS;
    const int c = (blk / NHEADS) % NCHUNK;
    const int b = blk / (NHEADS * NCHUNK);
    const int tid = threadIdx.x;
    const int t0 = c * Q;

    __shared__ __align__(16) u16 sXw[64 * CLDS];
    __shared__ __align__(16) u16 sB [64 * CLDS];
    __shared__ float scum[Q];
    __shared__ float sw[Q];

    chunk_cumsum(dtAb, b, t0, h, tid, scum);
    if (tid < Q) {
        float dt = dtb[(size_t)(b * SEQLEN + t0 + tid) * NHEADS + h];
        sw[tid] = __expf(scum[Q - 1] - scum[tid]) * dt;
    }
    if (tid == 0) pcum[(b * NHEADS + h) * NCHUNK + c] = scum[Q - 1];
    __syncthreads();

    {
        int r  = tid & 63;
        int sh = (tid >> 6) * 32;
#pragma unroll
        for (int g = 0; g < 4; g++) {
            float xv[8], bv[8];
#pragma unroll
            for (int i = 0; i < 8; i++) {
                int s = sh + g * 8 + i;
                size_t ga = (size_t)(b * SEQLEN + t0 + s) * CONV_DIM;
                xv[i] = sw[s] * xBC[ga + h * HEADDIM + r];
                bv[i] = xBC[ga + D_INNER + r];
            }
            uint4 px, pb;
            px.x = pack_bf16(xv[0], xv[1]); px.y = pack_bf16(xv[2], xv[3]);
            px.z = pack_bf16(xv[4], xv[5]); px.w = pack_bf16(xv[6], xv[7]);
            pb.x = pack_bf16(bv[0], bv[1]); pb.y = pack_bf16(bv[2], bv[3]);
            pb.z = pack_bf16(bv[4], bv[5]); pb.w = pack_bf16(bv[6], bv[7]);
            *(uint4*)&sXw[r * CLDS + sh + g * 8] = px;
            *(uint4*)&sB [r * CLDS + sh + g * 8] = pb;
        }
    }
    __syncthreads();

    const int lane = tid & 31, wid = tid >> 5;
    const int lm = lane & 15, lh = lane >> 4;
    v8f acc[4] = {};
#pragma unroll
    for (int ks = 0; ks < 2; ks++) {
        v16bf fa = frag_ld(&sXw[(wid * 16 + lm) * CLDS + ks * 32], lh);
#pragma unroll
        for (int j = 0; j < 4; j++) {
            v16bf fb = frag_ld(&sB[(j * 16 + lm) * CLDS + ks * 32], lh);
            acc[j] = wmma_bf16(fa, fb, acc[j]);
        }
    }
    float* tdst = tbuf + ((size_t)(b * NHEADS + h) * NCHUNK + c) * (HEADDIM * D_STATE);
#pragma unroll
    for (int j = 0; j < 4; j++)
#pragma unroll
        for (int r = 0; r < 8; r++) {
            int p = wid * 16 + lh * 8 + r;
            int n = j * 16 + lm;
            tdst[p * D_STATE + n] = acc[j][r];
        }
}

// Phase B: tiny sequential recurrence across chunks
__global__ __launch_bounds__(256) void state_recur_kernel(
    float* __restrict__ tbuf, const float* __restrict__ pcum)
{
    int bh = blockIdx.x;
    int tid = threadIdx.x;
    float s[16];
#pragma unroll
    for (int i = 0; i < 16; i++) s[i] = 0.0f;
    float* base = tbuf + (size_t)bh * NCHUNK * (HEADDIM * D_STATE);
    const float* pc = pcum + bh * NCHUNK;
    for (int c = 0; c < NCHUNK; c++) {
        float P = __expf(pc[c]);
        float* slot = base + (size_t)c * (HEADDIM * D_STATE);
#pragma unroll
        for (int i = 0; i < 16; i++) {
            int e = tid + i * 256;
            float T = slot[e];
            slot[e] = s[i];
            s[i] = s[i] * P + T;
        }
    }
}

// Phase C: Y = (mask ∘ (C·Bᵀ))·X + diag(exp(cumA))·(C·S_startᵀ) + D·x
__global__ __launch_bounds__(128) void chunk_output_kernel(
    const float* __restrict__ xBC, const float* __restrict__ dtb,
    const float* __restrict__ dtAb, const float* __restrict__ tbuf,
    const float* __restrict__ Dp, float* __restrict__ ybuf)
{
    const int blk = blockIdx.x;
    const int h = blk % NHEADS;
    const int c = (blk / NHEADS) % NCHUNK;
    const int b = blk / (NHEADS * NCHUNK);
    const int tid = threadIdx.x;
    const int t0 = c * Q;

    __shared__ __align__(16) u16 sC [64 * CLDS];
    __shared__ __align__(16) u16 sBt[64 * CLDS];
    __shared__ __align__(16) u16 sX [64 * CLDS];
    __shared__ __align__(16) u16 sS [64 * CLDS];
    __shared__ __align__(16) u16 sM [64 * CLDS];
    __shared__ float scum[Q];
    __shared__ float sdt[Q];

    chunk_cumsum(dtAb, b, t0, h, tid, scum);
    if (tid < Q) sdt[tid] = dtb[(size_t)(b * SEQLEN + t0 + tid) * NHEADS + h];
    __syncthreads();

    {
        int r  = tid & 63;
        int nh = (tid >> 6) * 32;
        size_t rowt = (size_t)(b * SEQLEN + t0 + r) * CONV_DIM;
        const float* ss = tbuf + ((size_t)(b * NHEADS + h) * NCHUNK + c) * (HEADDIM * D_STATE);
#pragma unroll
        for (int g = 0; g < 4; g++) {
            float cv[8], bv[8], xv[8], sv[8];
#pragma unroll
            for (int i = 0; i < 8; i++) {
                int q = nh + g * 8 + i;
                cv[i] = xBC[rowt + D_INNER + D_STATE + q];
                bv[i] = xBC[rowt + D_INNER + q];
                sv[i] = ss[r * D_STATE + q];
                xv[i] = xBC[(size_t)(b * SEQLEN + t0 + q) * CONV_DIM + h * HEADDIM + r];
            }
            uint4 pc4, pb4, px4, ps4;
            pc4.x = pack_bf16(cv[0], cv[1]); pc4.y = pack_bf16(cv[2], cv[3]);
            pc4.z = pack_bf16(cv[4], cv[5]); pc4.w = pack_bf16(cv[6], cv[7]);
            pb4.x = pack_bf16(bv[0], bv[1]); pb4.y = pack_bf16(bv[2], bv[3]);
            pb4.z = pack_bf16(bv[4], bv[5]); pb4.w = pack_bf16(bv[6], bv[7]);
            px4.x = pack_bf16(xv[0], xv[1]); px4.y = pack_bf16(xv[2], xv[3]);
            px4.z = pack_bf16(xv[4], xv[5]); px4.w = pack_bf16(xv[6], xv[7]);
            ps4.x = pack_bf16(sv[0], sv[1]); ps4.y = pack_bf16(sv[2], sv[3]);
            ps4.z = pack_bf16(sv[4], sv[5]); ps4.w = pack_bf16(sv[6], sv[7]);
            *(uint4*)&sC [r * CLDS + nh + g * 8] = pc4;
            *(uint4*)&sBt[r * CLDS + nh + g * 8] = pb4;
            *(uint4*)&sX [r * CLDS + nh + g * 8] = px4;
            *(uint4*)&sS [r * CLDS + nh + g * 8] = ps4;
        }
    }
    __syncthreads();

    const int lane = tid & 31, wid = tid >> 5;
    const int lm = lane & 15, lh = lane >> 4;

    v8f accG[4] = {};
    v8f accI[4] = {};
#pragma unroll
    for (int ks = 0; ks < 2; ks++) {
        v16bf fa = frag_ld(&sC[(wid * 16 + lm) * CLDS + ks * 32], lh);
#pragma unroll
        for (int j = 0; j < 4; j++) {
            v16bf fb = frag_ld(&sBt[(j * 16 + lm) * CLDS + ks * 32], lh);
            accG[j] = wmma_bf16(fa, fb, accG[j]);
        }
#pragma unroll
        for (int j = 0; j < 4; j++) {
            v16bf fs = frag_ld(&sS[(j * 16 + lm) * CLDS + ks * 32], lh);
            accI[j] = wmma_bf16(fa, fs, accI[j]);
        }
    }

#pragma unroll
    for (int j = 0; j < 4; j++)
#pragma unroll
        for (int r = 0; r < 8; r++) {
            int t = wid * 16 + lh * 8 + r;
            int s = j * 16 + lm;
            float f = (s <= t) ? __expf(scum[t] - scum[s]) * sdt[s] : 0.0f;
            sM[t * CLDS + s] = bf16_bits(accG[j][r] * f);
        }
    __syncthreads();

    v8f accY[4] = {};
#pragma unroll
    for (int ks = 0; ks < 2; ks++) {
        v16bf fa = frag_ld(&sM[(wid * 16 + lm) * CLDS + ks * 32], lh);
#pragma unroll
        for (int j = 0; j < 4; j++) {
            v16bf fb = frag_ld(&sX[(j * 16 + lm) * CLDS + ks * 32], lh);
            accY[j] = wmma_bf16(fa, fb, accY[j]);
        }
    }

    float Dh = Dp[h];
#pragma unroll
    for (int j = 0; j < 4; j++)
#pragma unroll
        for (int r = 0; r < 8; r++) {
            int t = wid * 16 + lh * 8 + r;
            int p = j * 16 + lm;
            float xv = xBC[(size_t)(b * SEQLEN + t0 + t) * CONV_DIM + h * HEADDIM + p];
            float y = accY[j][r] + __expf(scum[t]) * accI[j][r] + Dh * xv;
            ybuf[(size_t)(b * SEQLEN + t0 + t) * D_INNER + h * HEADDIM + p] = y;
        }
}

// =============== gate + RMSNorm; emits bf16 for the out-proj GEMM ===============
__global__ __launch_bounds__(256) void gatenorm_kernel(
    const float* __restrict__ zxbcdt, const float* __restrict__ norm_w,
    const float* __restrict__ ybuf, u16* __restrict__ ybf)
{
    int row = blockIdx.x;
    int t = threadIdx.x;
    __shared__ float red[256];
    float g[4];
    float ss = 0.0f;
#pragma unroll
    for (int i = 0; i < 4; i++) {
        int cc = t + i * 256;
        float y = ybuf[(size_t)row * D_INNER + cc];
        float z = zxbcdt[(size_t)row * D_IN_PROJ + cc];
        float gi = y * (z / (1.0f + __expf(-z)));
        g[i] = gi;
        ss += gi * gi;
    }
    red[t] = ss;
    __syncthreads();
    for (int off = 128; off > 0; off >>= 1) {
        if (t < off) red[t] += red[t + off];
        __syncthreads();
    }
    float rstd = rsqrtf(red[0] / (float)D_INNER + 1e-5f);
#pragma unroll
    for (int i = 0; i < 4; i++) {
        int cc = t + i * 256;
        ybf[(size_t)row * D_INNER + cc] = bf16_bits(g[i] * rstd * norm_w[cc]);
    }
}

// =============== launch ===============
extern "C" void kernel_launch(void* const* d_in, const int* in_sizes, int n_in,
                              void* d_out, int out_size, void* d_ws, size_t ws_size,
                              hipStream_t stream)
{
    const float* x        = (const float*)d_in[0];
    const float* W_in     = (const float*)d_in[1];
    const float* conv_w   = (const float*)d_in[2];
    const float* conv_b   = (const float*)d_in[3];
    const float* dt_bias  = (const float*)d_in[4];
    const float* A_log    = (const float*)d_in[5];
    const float* Dp       = (const float*)d_in[6];
    const float* norm_w   = (const float*)d_in[7];
    const float* W_out    = (const float*)d_in[8];
    float* out = (float*)d_out;

    float* ws = (float*)d_ws;
    float* zxbcdt = ws;            size_t off = (size_t)NTOK * D_IN_PROJ;
    float* xBC    = ws + off;      off += (size_t)NTOK * CONV_DIM;
    float* dtb    = ws + off;      off += (size_t)NTOK * NHEADS;
    float* dtAb   = ws + off;      off += (size_t)NTOK * NHEADS;
    float* tbuf   = ws + off;      off += (size_t)BATCH * NHEADS * NCHUNK * HEADDIM * D_STATE;
    float* pcum   = ws + off;      off += (size_t)BATCH * NHEADS * NCHUNK;
    float* ybuf   = ws + off;      off += (size_t)NTOK * D_INNER;
    // bf16 regions (u16), carved from the float workspace
    u16* xbf   = (u16*)(ws + off); off += (size_t)NTOK * D_MODEL / 2;
    u16* ybf   = (u16*)(ws + off); off += (size_t)NTOK * D_INNER / 2;
    u16* Wtin  = (u16*)(ws + off); off += (size_t)D_IN_PROJ * D_MODEL / 2;
    u16* Wtout = (u16*)(ws + off);

    // 0) one-shot bf16 pre-conversion of A matrices and transposed weights
    f32_to_bf16_kernel<<<(NTOK * D_MODEL / 8 + 255) / 256, 256, 0, stream>>>(
        x, xbf, NTOK * D_MODEL / 8);
    transpose_bf16_kernel<<<dim3((D_IN_PROJ + 31) / 32, (D_MODEL + 31) / 32), 256, 0, stream>>>(
        W_in, Wtin, D_MODEL, D_IN_PROJ);
    transpose_bf16_kernel<<<dim3((D_MODEL + 31) / 32, (D_INNER + 31) / 32), 256, 0, stream>>>(
        W_out, Wtout, D_INNER, D_MODEL);

    // 1) in-proj GEMM: 8192 x 2192 x 512
    dim3 g1((D_IN_PROJ + BN - 1) / BN, NTOK / BM);
    gemm_bf16wmma<<<g1, 256, 0, stream>>>(xbf, Wtin, zxbcdt, NTOK, D_IN_PROJ, D_MODEL);

    // 2) conv1d + SiLU
    conv_silu_kernel<<<(NTOK * CONV_DIM + 255) / 256, 256, 0, stream>>>(
        zxbcdt, conv_w, conv_b, xBC);

    // 3) dt / dtA
    dt_kernel<<<(NTOK * NHEADS + 255) / 256, 256, 0, stream>>>(
        zxbcdt, dt_bias, A_log, dtb, dtAb);

    // 4) chunked SSD scan
    chunk_state_kernel<<<BATCH * NCHUNK * NHEADS, 128, 0, stream>>>(
        xBC, dtb, dtAb, tbuf, pcum);
    state_recur_kernel<<<BATCH * NHEADS, 256, 0, stream>>>(tbuf, pcum);
    chunk_output_kernel<<<BATCH * NCHUNK * NHEADS, 128, 0, stream>>>(
        xBC, dtb, dtAb, tbuf, Dp, ybuf);

    // 5) gate + RMSNorm -> bf16
    gatenorm_kernel<<<NTOK, 256, 0, stream>>>(zxbcdt, norm_w, ybuf, ybf);

    // 6) out-proj GEMM: 8192 x 512 x 1024
    dim3 g2(D_MODEL / BN, NTOK / BM);
    gemm_bf16wmma<<<g2, 256, 0, stream>>>(ybf, Wtout, out, NTOK, D_MODEL, D_INNER);
}
